// CrossScanSS2D_58892591563289
// MI455X (gfx1250) — matrix operand order repaired and checked by reading, not compiled, not run on hardware
//
#include <hip/hip_runtime.h>

// ---------------------------------------------------------------------------
// CrossScanSS2D for MI455X (gfx1250, wave32, WMMA + TDM)
//
// Shapes: B=16, H=W=64, D=384, T=B*H*W=65536 tokens.
// Phase 1 (matrix cores): retain = sigmoid(-(X@Wd^T+bd)), val = X@Wv^T
//   - 32-token M-tile/block: every B fragment feeds 2 M-subtiles (halves L2)
//   - 4 independent WMMA chains/wave covers the WMMA->WMMA hazard window
//   - X staged by the Tensor Data Mover in 16-row chunks
// Phase 2 (bandwidth): 4 first-order recurrences over L=64, merge (bf16 r/v)
// Phase 3: LayerNorm + residual
// ---------------------------------------------------------------------------

typedef __attribute__((ext_vector_type(16))) __bf16 v16bf;
typedef __attribute__((ext_vector_type(8)))  float  v8f;
typedef unsigned int u32x4 __attribute__((ext_vector_type(4)));
typedef int          i32x8 __attribute__((ext_vector_type(8)));
typedef int          i32x4 __attribute__((ext_vector_type(4)));

#define Bsz 16
#define Hd  64
#define Wd_ 64
#define Dm  384
#define Tt  (Bsz * Hd * Wd_)      // 65536 tokens
#define KT  12                     // 384 / 32 K-steps
#define NT  24                     // 384 / 16 N-tiles

__device__ __forceinline__ unsigned short f2bf(float f) {
  unsigned int u = __float_as_uint(f);
  unsigned int r = u + 0x7FFFu + ((u >> 16) & 1u);   // round-to-nearest-even
  return (unsigned short)(r >> 16);
}
__device__ __forceinline__ float bf2f(unsigned short h) {
  return __uint_as_float(((unsigned int)h) << 16);
}

// ---------------------------------------------------------------------------
// Kernel 0: pack fp32 weights into bf16 WMMA B-fragment order.
// B-matrix (KxN=32x16, 16-bit): lane<16 -> N=lane, K=0..7,16..23; lane>=16 -> K=8..15,24..31
// ---------------------------------------------------------------------------
__global__ void ss2d_pack_w(const float* __restrict__ Wd,
                            const float* __restrict__ Wv,
                            unsigned short* __restrict__ PWd,
                            unsigned short* __restrict__ PWv) {
  int tid = blockIdx.x * blockDim.x + threadIdx.x;
  if (tid >= 2 * Dm * Dm) return;
  int mat  = tid / (Dm * Dm);
  int r    = tid % (Dm * Dm);
  int f    = r / 512;              // frag = ntile*KT + kstep
  int lane = (r % 512) / 16;
  int e    = r % 16;
  int ntile = f / KT, kstep = f % KT;
  int n = ntile * 16 + (lane & 15);
  int k = kstep * 32 + (e >> 3) * 16 + (lane >> 4) * 8 + (e & 7);
  const float* W = mat ? Wv : Wd;
  unsigned short* P = mat ? PWv : PWd;
  P[(size_t)f * 512 + lane * 16 + e] = f2bf(W[n * Dm + k]);  // Bmat[k][n]=W[n][k]
}

// ---------------------------------------------------------------------------
// Kernel 1: dual GEMM (retain + value) via v_wmma_f32_16x16x32_bf16.
// Block = 256 threads (8 waves), 32-token M-tile (2 subtiles of 16).
// Waves 0-3: retain (Wd); waves 4-7: value (Wv). Each wave: 6 N-tiles as 3
// pairs; per pair, 4 independent accumulator chains (2 Msub x 2 Ntile).
// ---------------------------------------------------------------------------
__global__ void __launch_bounds__(256, 1)
ss2d_gemm(const float* __restrict__ X,
          const unsigned short* __restrict__ PWd,
          const unsigned short* __restrict__ PWv,
          const float* __restrict__ bd,
          unsigned short* __restrict__ Rb,
          unsigned short* __restrict__ Vb) {
  // LDS: [0,24576) fp32 TDM staging (16 rows); [24576,49152) bf16 A fragments.
  __shared__ __align__(32) unsigned char smem[24576 + 2 * KT * 32 * 16 * 2];
  float*          sX = (float*)smem;
  unsigned short* sA = (unsigned short*)(smem + 24576);

  const int tid  = threadIdx.x;
  const int lane = tid & 31;
  const int wave = tid >> 5;
  const int blk  = blockIdx.x;                  // 32-token M-tile index

  // --- Stage X: two 16-row chunks, TDM -> LDS staging -> bf16 A-fragments ---
  for (int c = 0; c < 2; ++c) {
    if (tid < 32) {
      unsigned long long ga =
          (unsigned long long)(X + ((size_t)blk * 32 + c * 16) * Dm);
      u32x4 g0;
      g0[0] = 1u;                                          // count=1, user mode
      g0[1] = 0u;                                          // lds_addr = 0
      g0[2] = (unsigned)(ga & 0xFFFFFFFFu);                // global_addr[31:0]
      g0[3] = (unsigned)((ga >> 32) & 0x1FFFFFFu) | (2u << 30);  // type=2
      i32x8 g1;
      g1[0] = (int)(2u << 16);                             // data_size=4B
      g1[1] = (int)((Dm & 0xFFFF) << 16);                  // tensor_dim0 lo16
      g1[2] = 0;
      g1[3] = (int)(1u | ((unsigned)Dm << 16));            // dim1 hi16, tile_dim0
      g1[4] = 16;                                          // tile_dim1=16
      g1[5] = Dm;                                          // dim0 stride lo32
      g1[6] = 0;
      g1[7] = 0;
      i32x4 z4 = {0, 0, 0, 0};
      i32x8 z8 = {0, 0, 0, 0, 0, 0, 0, 0};
      __builtin_amdgcn_tensor_load_to_lds(g0, g1, z4, z4, z8, 0);
      __builtin_amdgcn_s_wait_tensorcnt(0);
    }
    __syncthreads();                            // staged data visible to all
    #pragma unroll
    for (int it = 0; it < 24; ++it) {
      int idx = tid + it * 256;                 // 0 .. 6143
      int m = idx / Dm, k = idx % Dm;
      float x = sX[m * Dm + k];
      int kstep = k >> 5, kk = k & 31;
      int hi = (kk >> 3) & 1;
      int ln = m + 16 * hi;
      int kb = kk - 8 * hi;
      int e  = (kb < 8) ? kb : (kb - 8);
      sA[((c * KT + kstep) * 32 + ln) * 16 + e] = f2bf(x);
    }
    __syncthreads();                            // staging free for next chunk
  }

  const int mat = wave >> 2;                    // waves 0-3: Wd, 4-7: Wv
  const unsigned short* Pbase = mat ? PWv : PWd;
  const int nt0 = (wave & 3) * 6;               // this wave's first N-tile

  for (int jp = 0; jp < 3; ++jp) {
    int ntA = nt0 + jp * 2;
    int ntB = ntA + 1;
    const unsigned short* pwA = Pbase + ((size_t)ntA * KT * 512 + lane * 16);
    const unsigned short* pwB = Pbase + ((size_t)ntB * KT * 512 + lane * 16);

    // Depth-2 rolling B prefetch; 4 independent WMMA chains.
    v16bf a0 = *reinterpret_cast<const v16bf*>(pwA);
    v16bf b0 = *reinterpret_cast<const v16bf*>(pwB);
    v16bf a1 = *reinterpret_cast<const v16bf*>(pwA + 512);
    v16bf b1 = *reinterpret_cast<const v16bf*>(pwB + 512);
    v8f acc00 = {};  // [msub0][ntA]
    v8f acc01 = {};  // [msub0][ntB]
    v8f acc10 = {};  // [msub1][ntA]
    v8f acc11 = {};  // [msub1][ntB]
    #pragma unroll
    for (int ks = 0; ks < KT; ++ks) {
      v16bf A0 = *reinterpret_cast<const v16bf*>(&sA[(ks * 32 + lane) * 16]);
      v16bf A1 = *reinterpret_cast<const v16bf*>(
          &sA[((KT + ks) * 32 + lane) * 16]);
      v16bf an = a1, bn = b1;
      if (ks + 2 < KT) {
        an = *reinterpret_cast<const v16bf*>(pwA + (size_t)(ks + 2) * 512);
        bn = *reinterpret_cast<const v16bf*>(pwB + (size_t)(ks + 2) * 512);
      }
      acc00 = __builtin_amdgcn_wmma_f32_16x16x32_bf16(
          false, A0, false, a0, (short)0, acc00, false, false);
      acc01 = __builtin_amdgcn_wmma_f32_16x16x32_bf16(
          false, A0, false, b0, (short)0, acc01, false, false);
      acc10 = __builtin_amdgcn_wmma_f32_16x16x32_bf16(
          false, A1, false, a0, (short)0, acc10, false, false);
      acc11 = __builtin_amdgcn_wmma_f32_16x16x32_bf16(
          false, A1, false, b0, (short)0, acc11, false, false);
      a0 = a1; a1 = an;
      b0 = b1; b1 = bn;
    }

    // C/D layout: vgpr g, lane l -> M = g + 8*(l>=16), N = l&15
    int mAdd = (lane >> 4) * 8;
    #pragma unroll
    for (int msub = 0; msub < 2; ++msub) {
      #pragma unroll
      for (int half = 0; half < 2; ++half) {
        int nt = half ? ntB : ntA;
        v8f acc = msub ? (half ? acc11 : acc10) : (half ? acc01 : acc00);
        int ng = nt * 16 + (lane & 15);
        if (mat == 0) {
          float bias = bd[ng];
          #pragma unroll
          for (int g = 0; g < 8; ++g) {
            int t = blk * 32 + msub * 16 + g + mAdd;
            float val = acc[g] + bias;
            float rt = 1.0f / (1.0f + __expf(val));  // exp(-softplus(val))
            Rb[(size_t)t * Dm + ng] = f2bf(rt);
          }
        } else {
          #pragma unroll
          for (int g = 0; g < 8; ++g) {
            int t = blk * 32 + msub * 16 + g + mAdd;
            Vb[(size_t)t * Dm + ng] = f2bf(acc[g]);
          }
        }
      }
    }
  }
}

// ---------------------------------------------------------------------------
// Kernel 2a: row scans (LR + RL). Block = 128 threads = 128 channels.
// grid = (B*H) * 3 channel-chunks. Token(w) = seq*64 + w.
// ---------------------------------------------------------------------------
__global__ void ss2d_scan_rows(const unsigned short* __restrict__ Rb,
                               const unsigned short* __restrict__ Vb,
                               const float* __restrict__ rs,
                               float* __restrict__ out) {
  __shared__ unsigned short sr[64 * 128];
  __shared__ unsigned short sv[64 * 128];
  const int tid = threadIdx.x;
  const int seq = blockIdx.x / 3;               // b*64 + h
  const int d   = (blockIdx.x % 3) * 128 + tid;
  const float s0 = rs[0 * Dm + d];
  const float s1 = rs[1 * Dm + d];

  #pragma unroll 4
  for (int w = 0; w < 64; ++w) {
    size_t idx = (size_t)(seq * 64 + w) * Dm + d;
    sr[w * 128 + tid] = Rb[idx];
    sv[w * 128 + tid] = Vb[idx];
  }
  __syncthreads();

  float fwd[64];
  float h = 0.0f;
  #pragma unroll
  for (int w = 0; w < 64; ++w) {
    float r = bf2f(sr[w * 128 + tid]);
    float v = bf2f(sv[w * 128 + tid]) * s0;
    h = r * h + (1.0f - r) * v;
    fwd[w] = h;
  }
  float hb = 0.0f;
  #pragma unroll
  for (int w = 63; w >= 0; --w) {
    float r = bf2f(sr[w * 128 + tid]);
    float v = bf2f(sv[w * 128 + tid]) * s1;
    hb = r * hb + (1.0f - r) * v;
    out[(size_t)(seq * 64 + w) * Dm + d] = 0.25f * (fwd[w] + hb);
  }
}

// ---------------------------------------------------------------------------
// Kernel 2b: column scans (TB + BT). Token(h) = b*4096 + h*64 + w.
// Read-modify-write out += 0.25*(y_tb + y_bt).
// ---------------------------------------------------------------------------
__global__ void ss2d_scan_cols(const unsigned short* __restrict__ Rb,
                               const unsigned short* __restrict__ Vb,
                               const float* __restrict__ rs,
                               float* __restrict__ out) {
  __shared__ unsigned short sr[64 * 128];
  __shared__ unsigned short sv[64 * 128];
  const int tid = threadIdx.x;
  const int cw  = blockIdx.x / 3;               // b*64 + w
  const int b   = cw / 64, w = cw % 64;
  const int d   = (blockIdx.x % 3) * 128 + tid;
  const float s2 = rs[2 * Dm + d];
  const float s3 = rs[3 * Dm + d];

  #pragma unroll 4
  for (int h = 0; h < 64; ++h) {
    size_t idx = (size_t)(b * 4096 + h * 64 + w) * Dm + d;
    sr[h * 128 + tid] = Rb[idx];
    sv[h * 128 + tid] = Vb[idx];
  }
  __syncthreads();

  float fwd[64];
  float hv = 0.0f;
  #pragma unroll
  for (int h = 0; h < 64; ++h) {
    float r = bf2f(sr[h * 128 + tid]);
    float v = bf2f(sv[h * 128 + tid]) * s2;
    hv = r * hv + (1.0f - r) * v;
    fwd[h] = hv;
  }
  float hb = 0.0f;
  #pragma unroll
  for (int h = 63; h >= 0; --h) {
    float r = bf2f(sr[h * 128 + tid]);
    float v = bf2f(sv[h * 128 + tid]) * s3;
    hb = r * hb + (1.0f - r) * v;
    size_t idx = (size_t)(b * 4096 + h * 64 + w) * Dm + d;
    out[idx] = out[idx] + 0.25f * (fwd[h] + hb);
  }
}

// ---------------------------------------------------------------------------
// Kernel 3: LayerNorm (biased var, eps=1e-5) + x*D_param residual, in place.
// ---------------------------------------------------------------------------
__global__ void ss2d_ln(float* __restrict__ out,
                        const float* __restrict__ X,
                        const float* __restrict__ gamma,
                        const float* __restrict__ beta,
                        const float* __restrict__ Dp) {
  __shared__ float red[32];
  const int tid = threadIdx.x;                  // 0..383
  const size_t idx = (size_t)blockIdx.x * Dm + tid;
  float val = out[idx];

  float s = val, s2 = val * val;
  #pragma unroll
  for (int o = 16; o > 0; o >>= 1) {
    s  += __shfl_down(s,  o, 32);
    s2 += __shfl_down(s2, o, 32);
  }
  int wv = tid >> 5;                            // 12 waves
  if ((tid & 31) == 0) { red[wv] = s; red[wv + 16] = s2; }
  __syncthreads();
  if (tid == 0) {
    float a = 0.0f, b2 = 0.0f;
    #pragma unroll
    for (int i = 0; i < 12; ++i) { a += red[i]; b2 += red[i + 16]; }
    float m = a * (1.0f / Dm);
    red[0] = m;
    red[1] = b2 * (1.0f / Dm) - m * m;
  }
  __syncthreads();
  float m = red[0], var = red[1];
  float y = (val - m) * rsqrtf(var + 1e-5f) * gamma[tid] + beta[tid];
  out[idx] = y + X[idx] * Dp[tid];
}

// ---------------------------------------------------------------------------
extern "C" void kernel_launch(void* const* d_in, const int* in_sizes, int n_in,
                              void* d_out, int out_size, void* d_ws, size_t ws_size,
                              hipStream_t stream) {
  const float* x    = (const float*)d_in[0];
  const float* Wd   = (const float*)d_in[1];
  const float* bd   = (const float*)d_in[2];
  const float* Wv   = (const float*)d_in[3];
  const float* rs   = (const float*)d_in[4];   // (4,1,1,D) -> [dir*D + d]
  const float* g    = (const float*)d_in[5];
  const float* bt   = (const float*)d_in[6];
  const float* Dp   = (const float*)d_in[7];
  float* out = (float*)d_out;

  // Workspace (ushorts): retain bf16 | value bf16 | packed Wd | packed Wv
  unsigned short* Rb  = (unsigned short*)d_ws;
  unsigned short* Vb  = Rb + (size_t)Tt * Dm;
  unsigned short* PWd = Vb + (size_t)Tt * Dm;
  unsigned short* PWv = PWd + (size_t)Dm * Dm;

  ss2d_pack_w<<<(2 * Dm * Dm + 255) / 256, 256, 0, stream>>>(Wd, Wv, PWd, PWv);
  ss2d_gemm<<<Tt / 32, 256, 0, stream>>>(x, PWd, PWv, bd, Rb, Vb);
  ss2d_scan_rows<<<Bsz * Hd * 3, 128, 0, stream>>>(Rb, Vb, rs, out);
  ss2d_scan_cols<<<Bsz * Wd_ * 3, 128, 0, stream>>>(Rb, Vb, rs, out);
  ss2d_ln<<<Tt, Dm, 0, stream>>>(out, x, g, bt, Dp);
}
